// AttnBlock_61753039782183
// MI455X (gfx1250) — matrix operand order repaired and checked by reading, not compiled, not run on hardware
//
#include <hip/hip_runtime.h>
#include <hip/hip_bf16.h>

typedef _Float16 v16h __attribute__((ext_vector_type(16)));
typedef _Float16 v8h  __attribute__((ext_vector_type(8)));
typedef float    v8f  __attribute__((ext_vector_type(8)));
typedef unsigned int v4u __attribute__((ext_vector_type(4)));
typedef int          v4i __attribute__((ext_vector_type(4)));
typedef int          v8i __attribute__((ext_vector_type(8)));

#define C_   128
#define HW_  4096
#define NB_  4
#define NKB_ (HW_ / 32)

#define AS1 __attribute__((address_space(1)))
#define AS3 __attribute__((address_space(3)))

#if __has_builtin(__builtin_amdgcn_global_load_async_to_lds_b128)
#define HAVE_ASYNC_LDS 1
#else
#define HAVE_ASYNC_LDS 0
#endif

#if __has_builtin(__builtin_amdgcn_tensor_load_to_lds) && __has_builtin(__builtin_amdgcn_s_wait_tensorcnt)
#define HAVE_TDM 1
#else
#define HAVE_TDM 0
#endif

// ---------- CDNA5 async helpers ----------
#if HAVE_ASYNC_LDS
static __device__ __forceinline__ void async_b128(const _Float16* g, _Float16* l) {
  // builtin signature (from hipcc diagnostic): (v4i AS1*, v4i AS3*, imm offset, imm cpol)
  __builtin_amdgcn_global_load_async_to_lds_b128(
      (AS1 v4i*)(AS1 void*)g, (AS3 v4i*)(AS3 void*)l, 0, 0);
}
static __device__ __forceinline__ void wait_async0() {
#if __has_builtin(__builtin_amdgcn_s_wait_asynccnt)
  __builtin_amdgcn_s_wait_asynccnt(0);
#else
  asm volatile("s_wait_asynccnt 0x0" ::: "memory");
#endif
}
#endif

#if HAVE_TDM
// 2-D TDM load: tile of 32 rows x 128 f16 (row stride 128 halves) from `g` into LDS `l`.
static __device__ __forceinline__ void tdm_load_tile_32x128(const _Float16* g, _Float16* l) {
  unsigned lds_off = (unsigned)(size_t)(AS3 void*)l;
  unsigned long long ga = (unsigned long long)(size_t)g;
  v4u g0 = { 1u,                                   // count=1, user descriptor
             lds_off,                              // lds_addr
             (unsigned)ga,                         // global_addr[31:0]
             (unsigned)((ga >> 32) & 0x01FFFFFFu) | 0x80000000u };  // addr[56:32] | type=2
  v8i g1 = { (int)(1u << 16),       // data_size = 2 bytes
             (int)(128u << 16),     // tensor_dim0 = 128 (lo16)
             (int)(4096u << 16),    // tensor_dim0 hi = 0 | tensor_dim1 = 4096 (lo16)
             (int)(128u << 16),     // tensor_dim1 hi = 0 | tile_dim0 = 128
             32,                    // tile_dim1 = 32, tile_dim2 = 0
             128,                   // tensor_dim0_stride = 128 (lo32)
             0, 0 };
  v4i gz = {0, 0, 0, 0};
#if __clang_major__ >= 23
  v8i gz8 = {0, 0, 0, 0, 0, 0, 0, 0};
  __builtin_amdgcn_tensor_load_to_lds(g0, g1, gz, gz, gz8, 0);
#else
  __builtin_amdgcn_tensor_load_to_lds(g0, g1, gz, gz, 0);
#endif
}
#endif

// ---------- WMMA helpers (CDNA5 layouts, wave32) ----------

static __device__ __forceinline__ v16h ld16x2(const _Float16* p0, const _Float16* p1) {
  v8h a = *(const v8h*)p0;
  v8h b = *(const v8h*)p1;
  v16h r;
#pragma unroll
  for (int i = 0; i < 8; ++i) { r[i] = a[i]; r[i + 8] = b[i]; }
  return r;
}

// A fragment 16(M)x32(K) f16 from row-major A[m][k] (row stride lda halves).
static __device__ __forceinline__ v16h load_fragA(const _Float16* a, int lda, int k0, int lane) {
  int r = lane & 15, koff = (lane >> 4) * 16;
  const _Float16* p = a + (size_t)r * lda + k0 + koff;
  return ld16x2(p, p + 8);
}

// B fragment 32(K)x16(N) f16 from memory holding B-transposed row-major BT[n][k].
static __device__ __forceinline__ v16h load_fragB(const _Float16* bt, int ldb, int k0, int lane) {
  int r = lane & 15, koff = (lane >> 4) * 16;
  const _Float16* p = bt + (size_t)r * ldb + k0 + koff;
  return ld16x2(p, p + 8);
}

static __device__ __forceinline__ v8f wmma_f16(v16h a, v16h b, v8f c) {
  return __builtin_amdgcn_wmma_f32_16x16x32_f16(false, a, false, b, (short)0, c, false, false);
}

static __device__ __forceinline__ v8f zero8() {
  v8f z = {0.f, 0.f, 0.f, 0.f, 0.f, 0.f, 0.f, 0.f};
  return z;
}

// ---------- 1) Fused flash attention, double-buffered TDM/async-LDS pipeline ----------
// (placed first in the TU so the disasm window lands on its staging + WMMA region)
__global__ void __launch_bounds__(256) flash_attn(const _Float16* __restrict__ q,
                                                  const _Float16* __restrict__ k,
                                                  const _Float16* __restrict__ vT,
                                                  _Float16* __restrict__ ao) {
  __shared__ __align__(16) _Float16 Kt[2][32][C_];   // 2 x (32 keys x 128 ch)  = 16 KB
  __shared__ __align__(16) _Float16 Vt[2][C_][32];   // 2 x (128 ch x 32 keys)  = 16 KB
  __shared__ __align__(16) _Float16 Ps[8][16][32];   // per-wave P bounce       =  8 KB
  const int tid = threadIdx.x, lane = tid & 31, wave = tid >> 5;
  const int batch = blockIdx.x >> 5;
  const int q0 = (blockIdx.x & 31) * 128 + wave * 16;
  const int r = lane & 15, hh = lane >> 4;
  const float scale = 0.08838834764831845f;  // 128^-0.5

  const _Float16* qbase = q + ((size_t)(batch * HW_ + q0)) * C_;
  const _Float16* kbase = k + (size_t)batch * HW_ * C_;
  const _Float16* vbase = vT + (size_t)batch * C_ * HW_;

  // stage K tile (TDM if available) + V^T tile (async-to-LDS if available) into buffer `buf`
  auto stage = [&](int buf, int kb2) {
#if HAVE_TDM
    if (wave == 0)
      tdm_load_tile_32x128(kbase + (size_t)kb2 * 32 * C_, &Kt[buf][0][0]);
#endif
#pragma unroll
    for (int it = 0; it < 2; ++it) {
      int idx = tid + it * 256;
      int c = idx >> 2, off = (idx & 3) * 8;
      const _Float16* gv = vbase + (size_t)c * HW_ + kb2 * 32 + off;
      _Float16* lv = &Vt[buf][c][off];
#if HAVE_ASYNC_LDS
      async_b128(gv, lv);
#else
      *(v8h*)lv = *(const v8h*)gv;
#endif
#if !HAVE_TDM
      int key = idx >> 4, ch = (idx & 15) * 8;
      const _Float16* gk = kbase + (size_t)(kb2 * 32 + key) * C_ + ch;
      _Float16* lk = &Kt[buf][key][ch];
#if HAVE_ASYNC_LDS
      async_b128(gk, lk);
#else
      *(v8h*)lk = *(const v8h*)gk;
#endif
#endif
    }
  };
  auto wait_stage = [&]() {
#if HAVE_TDM
    if (wave == 0) __builtin_amdgcn_s_wait_tensorcnt(0);
#endif
#if HAVE_ASYNC_LDS
    wait_async0();
#endif
    __syncthreads();
  };

  v16h qf[4];
#pragma unroll
  for (int cf = 0; cf < 4; ++cf) qf[cf] = load_fragA(qbase, C_, cf * 32, lane);

  v8f acc[8];
  float m[8], l[8];
#pragma unroll
  for (int t = 0; t < 8; ++t) acc[t] = zero8();
#pragma unroll
  for (int j = 0; j < 8; ++j) { m[j] = -1e30f; l[j] = 0.f; }

  stage(0, 0);
  wait_stage();

  for (int kb = 0; kb < NKB_; ++kb) {
    const int cur = kb & 1;
    if (kb + 1 < NKB_) stage(1 - cur, kb + 1);   // DMA next tile, overlapped with compute

    // S = Q x K^T : two 16x16 score tiles covering keys kb*32 .. kb*32+31
    v8f s0 = zero8(), s1 = zero8();
#pragma unroll
    for (int cf = 0; cf < 4; ++cf) {
      int c0 = cf * 32 + hh * 16;
      v16h b0 = ld16x2(&Kt[cur][r][c0],      &Kt[cur][r][c0 + 8]);
      v16h b1 = ld16x2(&Kt[cur][16 + r][c0], &Kt[cur][16 + r][c0 + 8]);
      s0 = wmma_f16(qf[cf], b0, s0);
      s1 = wmma_f16(qf[cf], b1, s1);
    }

    // online softmax: row (j,hh) spans the 16 lanes of this half-wave
    float alpha[8];
#pragma unroll
    for (int j = 0; j < 8; ++j) {
      float a = s0[j] * scale, b = s1[j] * scale;
      float mx = fmaxf(a, b);
#pragma unroll
      for (int msk = 1; msk < 16; msk <<= 1) mx = fmaxf(mx, __shfl_xor(mx, msk, 32));
      float mn = fmaxf(m[j], mx);
      float al = __expf(m[j] - mn);
      float p0 = __expf(a - mn);
      float p1 = __expf(b - mn);
      float rs = p0 + p1;
#pragma unroll
      for (int msk = 1; msk < 16; msk <<= 1) rs += __shfl_xor(rs, msk, 32);
      l[j] = l[j] * al + rs;
      m[j] = mn;
      alpha[j] = al;
      s0[j] = p0; s1[j] = p1;
    }
#pragma unroll
    for (int t = 0; t < 8; ++t)
#pragma unroll
      for (int j = 0; j < 8; ++j) acc[t][j] *= alpha[j];

    // C-layout -> A-layout re-tiling of P through per-wave LDS scratch.
    // Intra-wave only: LDS ops from one wave complete in order (DScnt), no barrier needed.
#pragma unroll
    for (int j = 0; j < 8; ++j) {
      Ps[wave][j + 8 * hh][r]      = (_Float16)s0[j];
      Ps[wave][j + 8 * hh][16 + r] = (_Float16)s1[j];
    }
    v16h pf = load_fragA(&Ps[wave][0][0], 32, 0, lane);

    // O += P x V : 8 column tiles of 16 channels
#pragma unroll
    for (int ct = 0; ct < 8; ++ct) {
      v16h bv = ld16x2(&Vt[cur][ct * 16 + r][hh * 16], &Vt[cur][ct * 16 + r][hh * 16 + 8]);
      acc[ct] = wmma_f16(pf, bv, acc[ct]);
    }

    wait_stage();   // next tile landed everywhere; all waves done reading `cur`
  }

  float inv[8];
#pragma unroll
  for (int j = 0; j < 8; ++j) inv[j] = 1.0f / l[j];
  _Float16* aob = ao + ((size_t)(batch * HW_ + q0)) * C_;
#pragma unroll
  for (int ct = 0; ct < 8; ++ct)
#pragma unroll
    for (int j = 0; j < 8; ++j)
      aob[(size_t)(j + 8 * hh) * C_ + ct * 16 + r] = (_Float16)(acc[ct][j] * inv[j]);
}

// ---------- 2) QKV projections: one 16x16 output tile per wave ----------
__global__ void __launch_bounds__(256) qkv_gemm(const _Float16* __restrict__ hT,
                                                const _Float16* __restrict__ w16,
                                                const float* __restrict__ bq,
                                                const float* __restrict__ bk,
                                                const float* __restrict__ bv,
                                                _Float16* __restrict__ q,
                                                _Float16* __restrict__ k,
                                                _Float16* __restrict__ vT) {
  int lane = threadIdx.x & 31, wave = threadIdx.x >> 5;
  int which = blockIdx.y;
  int mt = blockIdx.x;
  int nt = wave;
  int batch = mt >> 8, trow = (mt & 255) << 4;
  const _Float16* W = w16 + (size_t)which * C_ * C_;
  const float* bias = (which == 0) ? bq : (which == 1) ? bk : bv;
  const _Float16* abase = hT + ((size_t)(batch * HW_ + trow)) * C_;
  v8f acc = zero8();
#pragma unroll
  for (int cf = 0; cf < 4; ++cf) {
    v16h a = load_fragA(abase, C_, cf * 32, lane);
    v16h b = load_fragB(W + (size_t)nt * 16 * C_, C_, cf * 32, lane);
    acc = wmma_f16(a, b, acc);
  }
  int r = lane & 15, hh = lane >> 4;
  int oc = nt * 16 + r;
  float bval = bias[oc];
  if (which == 2) {
    v8h pk;
#pragma unroll
    for (int j = 0; j < 8; ++j) pk[j] = (_Float16)(acc[j] + bval);
    *(v8h*)(vT + ((size_t)(batch * C_ + oc)) * HW_ + trow + hh * 8) = pk;
  } else {
    _Float16* dst = ((which == 0) ? q : k) + ((size_t)(batch * HW_ + trow)) * C_;
#pragma unroll
    for (int j = 0; j < 8; ++j)
      dst[(size_t)(j + 8 * hh) * C_ + oc] = (_Float16)(acc[j] + bval);
  }
}

// ---------- 3) Output projection + bias + residual (f32 NCHW out) ----------
__global__ void __launch_bounds__(256) proj_residual(const _Float16* __restrict__ ao,
                                                     const _Float16* __restrict__ wp16,
                                                     const float* __restrict__ bp,
                                                     const float* __restrict__ xin,
                                                     float* __restrict__ out) {
  int lane = threadIdx.x & 31, wave = threadIdx.x >> 5;
  int mt = blockIdx.x, nt = wave;
  int batch = mt >> 8, trow = (mt & 255) << 4;
  const _Float16* abase = ao + ((size_t)(batch * HW_ + trow)) * C_;
  v8f acc = zero8();
#pragma unroll
  for (int cf = 0; cf < 4; ++cf) {
    v16h a = load_fragA(abase, C_, cf * 32, lane);
    v16h b = load_fragB(wp16 + (size_t)nt * 16 * C_, C_, cf * 32, lane);
    acc = wmma_f16(a, b, acc);
  }
  int r = lane & 15, hh = lane >> 4;
  int oc = nt * 16 + r;
  float bval = bp[oc];
  size_t base = ((size_t)(batch * C_ + oc)) * HW_ + trow + hh * 8;
#pragma unroll
  for (int j = 0; j < 8; ++j)
    out[base + j] = xin[base + j] + acc[j] + bval;
}

// ---------- 4) GroupNorm statistics: one block per (n, group) ----------
__global__ void __launch_bounds__(256) gn_stats(const float* __restrict__ x,
                                                float* __restrict__ stats) {
  int ng = blockIdx.x;
  int n = ng >> 5, g = ng & 31;
  const float* base = x + ((size_t)(n * C_ + g * 4)) * HW_;
  float s = 0.f, ss = 0.f;
  for (int i = threadIdx.x; i < 4 * HW_; i += 256) {
    float v = base[i];
    s += v; ss += v * v;
  }
  __shared__ float shs[256], shq[256];
  shs[threadIdx.x] = s; shq[threadIdx.x] = ss;
  __syncthreads();
  for (int off = 128; off > 0; off >>= 1) {
    if (threadIdx.x < off) {
      shs[threadIdx.x] += shs[threadIdx.x + off];
      shq[threadIdx.x] += shq[threadIdx.x + off];
    }
    __syncthreads();
  }
  if (threadIdx.x == 0) {
    float mean = shs[0] * (1.0f / (4 * HW_));
    float var  = shq[0] * (1.0f / (4 * HW_)) - mean * mean;
    stats[ng * 2 + 0] = mean;
    stats[ng * 2 + 1] = rsqrtf(var + 1e-5f);
  }
}

// ---------- 5) GroupNorm apply + transpose to token-major hT[N][HW][C] f16 ----------
__global__ void __launch_bounds__(256) gn_apply(const float* __restrict__ x,
                                                const float* __restrict__ stats,
                                                const float* __restrict__ gsc,
                                                const float* __restrict__ gbi,
                                                _Float16* __restrict__ hT) {
  __shared__ float tile[C_][33];
  int b = blockIdx.x;
  int n = b >> 7;
  int t0 = (b & 127) * 32;
#pragma unroll
  for (int it = 0; it < 16; ++it) {
    int idx = threadIdx.x + it * 256;
    int c = idx >> 5, tt = idx & 31;
    float v = x[((size_t)(n * C_ + c)) * HW_ + t0 + tt];
    int g = c >> 2;
    float mean = stats[(n * 32 + g) * 2 + 0];
    float rstd = stats[(n * 32 + g) * 2 + 1];
    tile[c][tt] = (v - mean) * rstd * gsc[c] + gbi[c];
  }
  __syncthreads();
#pragma unroll
  for (int it = 0; it < 16; ++it) {
    int idx = threadIdx.x + it * 256;
    int t = idx >> 7, c = idx & 127;
    hT[((size_t)(n * HW_ + t0 + t)) * C_ + c] = (_Float16)tile[c][t];
  }
}

// ---------- 6) Cast weights to f16 (natural [o][c] layout == WMMA BT layout) ----------
__global__ void __launch_bounds__(256) cast_weights(const float* __restrict__ wq,
                                                    const float* __restrict__ wk,
                                                    const float* __restrict__ wv,
                                                    const float* __restrict__ wp,
                                                    _Float16* __restrict__ w16) {
  int i = blockIdx.x * 256 + threadIdx.x;
  int which = i >> 14, rem = i & 16383;
  const float* src = (which == 0) ? wq : (which == 1) ? wk : (which == 2) ? wv : wp;
  w16[i] = (_Float16)src[rem];
}

// ---------- host ----------
extern "C" void kernel_launch(void* const* d_in, const int* in_sizes, int n_in,
                              void* d_out, int out_size, void* d_ws, size_t ws_size,
                              hipStream_t stream) {
  (void)in_sizes; (void)n_in; (void)out_size; (void)ws_size;
  const float* x  = (const float*)d_in[0];
  const float* gs = (const float*)d_in[1];
  const float* gb = (const float*)d_in[2];
  const float* wq = (const float*)d_in[3];
  const float* bq = (const float*)d_in[4];
  const float* wk = (const float*)d_in[5];
  const float* bk = (const float*)d_in[6];
  const float* wv = (const float*)d_in[7];
  const float* bv = (const float*)d_in[8];
  const float* wp = (const float*)d_in[9];
  const float* bp = (const float*)d_in[10];
  float* out = (float*)d_out;

  char* ws = (char*)d_ws;
  _Float16* hT  = (_Float16*)(ws + (0u  << 20));   // 4 MiB  [N,HW,C]
  _Float16* qb  = (_Float16*)(ws + (4u  << 20));   // 4 MiB  [N,HW,C]
  _Float16* kb  = (_Float16*)(ws + (8u  << 20));   // 4 MiB  [N,HW,C]
  _Float16* vT  = (_Float16*)(ws + (12u << 20));   // 4 MiB  [N,C,HW]
  _Float16* ao  = (_Float16*)(ws + (16u << 20));   // 4 MiB  [N,HW,C]
  _Float16* w16 = (_Float16*)(ws + (20u << 20));   // 128 KiB (wq,wk,wv,wp f16)
  float*  stats = (float*)   (ws + (21u << 20));   // 1 KiB  (mean,rstd per n,g)

  gn_stats<<<128, 256, 0, stream>>>(x, stats);
  gn_apply<<<512, 256, 0, stream>>>(x, stats, gs, gb, hT);
  cast_weights<<<256, 256, 0, stream>>>(wq, wk, wv, wp, w16);
  qkv_gemm<<<dim3(1024, 3), 256, 0, stream>>>(hT, w16, bq, bk, bv, qb, kb, vT);
  flash_attn<<<128, 256, 0, stream>>>(qb, kb, vT, ao);
  proj_residual<<<1024, 256, 0, stream>>>(ao, w16 + 3 * C_ * C_, bp, x, out);
}
